// NaiveAttentionModel_13434657702593
// MI455X (gfx1250) — matrix operand order repaired and checked by reading, not compiled, not run on hardware
//
#include <hip/hip_runtime.h>
#include <hip/hip_bf16.h>
#include <math.h>

typedef __attribute__((ext_vector_type(16))) _Float16 v16h;
typedef __attribute__((ext_vector_type(8)))  float    v8f;

#define N_LAYER 12
#define NHEAD   16
#define EDIM    1024
#define DIN     64
#define BATCH   2
#define KSEQ    512
#define LSEQ    1024
#define ROWS    (BATCH * LSEQ)   // 2048

// ---------------------------------------------------------------------------
// WMMA fragment helpers (gfx1250 wave32; this code's contiguous k order, same for A and B)
// A (16x32 f16): lane m = l%16; halves j: K = (l<16?0:16) + j
//   -> one contiguous 16-half (32B) run at p.
// B (32x16 f16): lane n = l%16; halves j: K = (l<16?0:16) + j
//   -> one contiguous 16-half (32B) run at p.
// C/D (16x16 f32): lane n = l%16; float r -> row M = r + (l<16?0:8).
// ---------------------------------------------------------------------------
__device__ __forceinline__ v16h load_a_frag(const _Float16* p) {
  v16h r;
  reinterpret_cast<float4*>(&r)[0] = *reinterpret_cast<const float4*>(p);
  reinterpret_cast<float4*>(&r)[1] = *reinterpret_cast<const float4*>(p + 8);
  return r;
}
__device__ __forceinline__ v16h load_b_frag(const _Float16* p) {
  v16h r;
  reinterpret_cast<float4*>(&r)[0] = *reinterpret_cast<const float4*>(p);
  reinterpret_cast<float4*>(&r)[1] = *reinterpret_cast<const float4*>(p + 8);
  return r;
}
__device__ __forceinline__ v8f wmma32(v16h a, v16h b, v8f c) {
  return __builtin_amdgcn_wmma_f32_16x16x32_f16(false, a, false, b, (short)0, c,
                                                false, false);
}

// ---------------------------------------------------------------------------
// Embedding: zs = interleave(x_i, [y_i,0...]) ; x = zs @ w_in + b_in
// ---------------------------------------------------------------------------
__global__ __launch_bounds__(256) void embed_kernel(
    const float* __restrict__ xs, const float* __restrict__ ys,
    const float* __restrict__ w_in, const float* __restrict__ b_in,
    float* __restrict__ x) {
  __shared__ float sz[DIN];
  const int row = blockIdx.x;
  const int b   = row >> 10;
  const int lr  = row & (LSEQ - 1);
  const int k   = lr >> 1;
  const bool odd = lr & 1;
  const int tid = threadIdx.x;
  if (tid < DIN) {
    if (!odd) sz[tid] = xs[(b * KSEQ + k) * DIN + tid];
    else      sz[tid] = (tid == 0) ? ys[b * KSEQ + k] : 0.0f;
  }
  __syncthreads();
  for (int c = tid; c < EDIM; c += 256) {
    float acc = b_in[c];
#pragma unroll 8
    for (int kk = 0; kk < DIN; ++kk) acc += sz[kk] * w_in[kk * EDIM + c];
    x[row * EDIM + c] = acc;
  }
}

// ---------------------------------------------------------------------------
// LayerNorm (biased variance, eps=1e-5), output f16 for WMMA A-operand
// ---------------------------------------------------------------------------
__global__ __launch_bounds__(256) void ln_kernel(
    const float* __restrict__ x, const float* __restrict__ g,
    const float* __restrict__ bta, _Float16* __restrict__ out) {
  __shared__ float rs[256], rq[256];
  const int row = blockIdx.x, tid = threadIdx.x;
  const float* xr = x + row * EDIM;
  float s = 0.f, q = 0.f;
  for (int c = tid; c < EDIM; c += 256) { float v = xr[c]; s += v; q += v * v; }
  rs[tid] = s; rq[tid] = q;
  __syncthreads();
  for (int st = 128; st > 0; st >>= 1) {
    if (tid < st) { rs[tid] += rs[tid + st]; rq[tid] += rq[tid + st]; }
    __syncthreads();
  }
  const float mean = rs[0] * (1.0f / EDIM);
  const float var  = rq[0] * (1.0f / EDIM) - mean * mean;
  const float rstd = rsqrtf(var + 1e-5f);
  for (int c = tid; c < EDIM; c += 256)
    out[row * EDIM + c] = (_Float16)((xr[c] - mean) * rstd * g[c] + bta[c]);
}

// ---------------------------------------------------------------------------
// Generic WMMA GEMM: C[M,N] = A_f16[M,K] * B_f32[K,N] + bias (+res, gelu)
// 256 threads = 8 waves (4 along M x 2 along N); tile 64x128; K-step 32.
// B staged through double-buffered LDS, transposed+converted to f16
// ([N][K], stride 40 halves -> 16B-aligned ds_load_b128 fragment reads).
// Staging is split load-phase / convert-phase so the 4 global_load_b128 can
// issue as one clause; uniform k0*N stays scalar (SALU), per-thread offsets
// are 32-bit.
// flags: 1=residual add (Res), 2=exact GELU, 4=write f32 (Of), 8=write f16 (Oh)
// ---------------------------------------------------------------------------
__global__ __launch_bounds__(256) void gemm_kernel(
    const _Float16* __restrict__ A, const float* __restrict__ Bw,
    const float* __restrict__ bias, const float* __restrict__ Res,
    float* __restrict__ Of, _Float16* __restrict__ Oh,
    int M, int N, int K, int flags) {
  __shared__ _Float16 sB[2][128 * 40];
  const int tid  = threadIdx.x;
  const int lane = tid & 31;
  const int w    = tid >> 5;
  const int wm   = w & 3;        // 0..3  (M sub-tile)
  const int wn   = w >> 2;       // 0..1  (N sub-tile)
  const int m0   = blockIdx.y * 64;
  const int n0   = blockIdx.x * 128;
  const int l15  = lane & 15;
  const int hi8  = (lane < 16) ? 0 : 8;

  // staging assignment: thread -> (row skk in 0..7, col group sc4 in 0..124)
  const int skk = tid >> 5;
  const int sc4 = (tid & 31) << 2;
  const int sroff = skk * N + n0 + sc4;        // 32-bit per-thread offset

  const v8f zf = {0.f, 0.f, 0.f, 0.f, 0.f, 0.f, 0.f, 0.f};
  v8f acc[4] = {zf, zf, zf, zf};

  const int arow = m0 + wm * 16 + l15;
  const _Float16* Abase = A + arow * K + ((lane < 16) ? 0 : 16);

  // stage one 32xN128 tile (rows k0s..k0s+31) into sB[buf]
  auto stage = [&](int k0s, int buf) {
    const float* p = Bw + (size_t)k0s * N + sroff;   // scalar mul, vector add
    const int kpf = (k0s + 32 < K) ? 32 : 0;         // uniform, branch-free
    __builtin_prefetch(p + (size_t)kpf * N, 0, 1);   // global_prefetch_b8
    float4 v[4];
#pragma unroll
    for (int i = 0; i < 4; ++i)                      // load phase (clause)
      v[i] = *reinterpret_cast<const float4*>(p + i * 8 * N);
    _Float16* dst0 = &sB[buf][sc4 * 40 + skk];
#pragma unroll
    for (int i = 0; i < 4; ++i) {                    // convert/store phase
      _Float16* dst = dst0 + 8 * i;
      dst[0]   = (_Float16)v[i].x;
      dst[40]  = (_Float16)v[i].y;
      dst[80]  = (_Float16)v[i].z;
      dst[120] = (_Float16)v[i].w;
    }
  };

  stage(0, 0);
  int buf = 0;
  for (int k0 = 0; k0 < K; k0 += 32) {
    __syncthreads();
    const v16h a = load_a_frag(Abase + k0);
    const _Float16* bbase =
        &sB[buf][(wn * 64 + l15) * 40 + ((lane < 16) ? 0 : 16)];
    const v16h b0 = load_b_frag(bbase);
    const v16h b1 = load_b_frag(bbase + 16 * 40);
    const v16h b2 = load_b_frag(bbase + 32 * 40);
    const v16h b3 = load_b_frag(bbase + 48 * 40);
    if (k0 + 32 < K) stage(k0 + 32, buf ^ 1);  // overlap with WMMA below
    acc[0] = wmma32(a, b0, acc[0]);
    acc[1] = wmma32(a, b1, acc[1]);
    acc[2] = wmma32(a, b2, acc[2]);
    acc[3] = wmma32(a, b3, acc[3]);
    buf ^= 1;
  }

  // epilogue (32-bit strength-reduced offsets)
#pragma unroll
  for (int nt = 0; nt < 4; ++nt) {
    const int col = n0 + wn * 64 + nt * 16 + l15;
    const float bv = bias[col];
    int off = (m0 + wm * 16 + hi8) * N + col;
#pragma unroll
    for (int r = 0; r < 8; ++r) {
      float v = acc[nt][r] + bv;
      if (flags & 2) v = 0.5f * v * (1.0f + erff(v * 0.70710678118654752f));
      if (flags & 1) v += Res[off];
      if (flags & 4) Of[off] = v;
      if (flags & 8) Oh[off] = (_Float16)v;
      off += N;
    }
  }
}

// ---------------------------------------------------------------------------
// Flash attention: one wave handles 16 query rows of one (batch, head).
// qkv layout: [row][c*E + h*64 + d] f16, row stride 3E. Causal, scale 1/8.
// ---------------------------------------------------------------------------
__global__ __launch_bounds__(32) void attn_kernel(
    const _Float16* __restrict__ qkv, _Float16* __restrict__ o16) {
  __shared__ _Float16 sP[16 * 32];    // P tile staging (C-layout -> A-layout)
  __shared__ _Float16 sVt[64 * 40];   // V^T block: [d][key], stride 40 halves
  const int lane = threadIdx.x;
  const int qblk = blockIdx.x & 63;
  const int h    = (blockIdx.x >> 6) & 15;
  const int b    = blockIdx.x >> 10;
  const int rowbase = b * LSEQ;
  const int qm0  = qblk * 16;
  const int l15  = lane & 15;
  const int hi8  = (lane < 16) ? 0 : 8;

  // Q A-fragments for both d-chunks (d=0..31, 32..63)
  const _Float16* qp = qkv + (rowbase + qm0 + l15) * 3072 + h * 64 +
                       ((lane < 16) ? 0 : 16);
  const v16h aq0 = load_a_frag(qp);
  const v16h aq1 = load_a_frag(qp + 32);

  const v8f zf = {0.f, 0.f, 0.f, 0.f, 0.f, 0.f, 0.f, 0.f};
  v8f o[4] = {zf, zf, zf, zf};
  float mrun[8], lrun[8];
#pragma unroll
  for (int r = 0; r < 8; ++r) { mrun[r] = -1e30f; lrun[r] = 0.f; }

  const int nkb = (qm0 + 47) >> 5;   // 32-key blocks covering keys <= qm0+15
  for (int kb = 0; kb < nkb; ++kb) {
    const int k0 = kb * 32;
    // ---- S = Q K^T for two 16-key subtiles -------------------------------
    v8f s0 = zf, s1 = zf;
    const _Float16* kp0 = qkv + (rowbase + k0 + l15) * 3072 + EDIM + h * 64 +
                          ((lane < 16) ? 0 : 16);
    s0 = wmma32(aq0, load_b_frag(kp0), s0);
    s0 = wmma32(aq1, load_b_frag(kp0 + 32), s0);
    const _Float16* kp1 = kp0 + 16 * 3072;
    s1 = wmma32(aq0, load_b_frag(kp1), s1);
    s1 = wmma32(aq1, load_b_frag(kp1 + 32), s1);

    // ---- causal mask + online softmax ------------------------------------
    const int key0 = k0 + l15, key1 = k0 + 16 + l15;
    float bm[8];
#pragma unroll
    for (int r = 0; r < 8; ++r) {
      const int qi = qm0 + r + hi8;
      const float v0 = (key0 <= qi) ? s0[r] * 0.125f : -1e30f;
      const float v1 = (key1 <= qi) ? s1[r] * 0.125f : -1e30f;
      s0[r] = v0; s1[r] = v1;
      bm[r] = fmaxf(v0, v1);
    }
#pragma unroll
    for (int r = 0; r < 8; ++r) {
#pragma unroll
      for (int off = 1; off < 16; off <<= 1)
        bm[r] = fmaxf(bm[r], __shfl_xor(bm[r], off, 16));
    }
#pragma unroll
    for (int r = 0; r < 8; ++r) {
      const float mnew = fmaxf(mrun[r], bm[r]);
      const float sc = __expf(mrun[r] - mnew);
      mrun[r] = mnew;
      const float p0 = __expf(s0[r] - mnew);
      const float p1 = __expf(s1[r] - mnew);
      float rsum = p0 + p1;
#pragma unroll
      for (int off = 1; off < 16; off <<= 1)
        rsum += __shfl_xor(rsum, off, 16);
      lrun[r] = lrun[r] * sc + rsum;
#pragma unroll
      for (int t = 0; t < 4; ++t) o[t][r] *= sc;
      sP[(r + hi8) * 32 + l15]      = (_Float16)p0;
      sP[(r + hi8) * 32 + 16 + l15] = (_Float16)p1;
    }

    // ---- stage V block transposed: sVt[d][key] ---------------------------
    {
      const _Float16* vp = qkv + (rowbase + k0 + lane) * 3072 + 2 * EDIM +
                           h * 64;
#pragma unroll
      for (int c8 = 0; c8 < 8; ++c8) {
        union { float4 f; _Float16 hh[8]; } raw;
        raw.f = *reinterpret_cast<const float4*>(vp + c8 * 8);
#pragma unroll
        for (int j = 0; j < 8; ++j) sVt[(c8 * 8 + j) * 40 + lane] = raw.hh[j];
      }
    }
    __syncthreads();

    // ---- O += P V --------------------------------------------------------
    const v16h ap = load_a_frag(sP + l15 * 32 + ((lane < 16) ? 0 : 16));
#pragma unroll
    for (int t = 0; t < 4; ++t) {
      const _Float16* bp = sVt + (t * 16 + l15) * 40 + ((lane < 16) ? 0 : 16);
      o[t] = wmma32(ap, load_b_frag(bp), o[t]);
    }
    __syncthreads();
  }

  // ---- normalize + store (o16[row][h*64 + d]) ----------------------------
#pragma unroll
  for (int r = 0; r < 8; ++r) {
    const float inv = 1.0f / lrun[r];
    const int off0 = (rowbase + qm0 + r + hi8) * EDIM + h * 64 + l15;
#pragma unroll
    for (int t = 0; t < 4; ++t)
      o16[off0 + t * 16] = (_Float16)(o[t][r] * inv);
  }
}

// ---------------------------------------------------------------------------
// Head: preds[b,k] = dot(x[b*L + 2k], w_out) + b_out
// ---------------------------------------------------------------------------
__global__ __launch_bounds__(256) void head_kernel(
    const float* __restrict__ x, const float* __restrict__ w_out,
    const float* __restrict__ b_out, float* __restrict__ out) {
  __shared__ float rs[256];
  const int i = blockIdx.x;          // b*KSEQ + k
  const int b = i >> 9, k = i & 511;
  const int row = b * LSEQ + 2 * k;
  const int tid = threadIdx.x;
  const float* xr = x + row * EDIM;
  float s = 0.f;
  for (int c = tid; c < EDIM; c += 256) s += xr[c] * w_out[c];
  rs[tid] = s;
  __syncthreads();
  for (int st = 128; st > 0; st >>= 1) {
    if (tid < st) rs[tid] += rs[tid + st];
    __syncthreads();
  }
  if (tid == 0) out[i] = rs[0] + b_out[0];
}

// ---------------------------------------------------------------------------
// Host launcher
// ---------------------------------------------------------------------------
extern "C" void kernel_launch(void* const* d_in, const int* in_sizes, int n_in,
                              void* d_out, int out_size, void* d_ws,
                              size_t ws_size, hipStream_t stream) {
  (void)in_sizes; (void)n_in; (void)out_size; (void)ws_size;
  const float* xs     = (const float*)d_in[0];
  const float* ys     = (const float*)d_in[1];
  const float* w_in   = (const float*)d_in[2];
  const float* b_in   = (const float*)d_in[3];
  const float* qkv_w  = (const float*)d_in[4];
  const float* qkv_b  = (const float*)d_in[5];
  const float* proj_w = (const float*)d_in[6];
  const float* proj_b = (const float*)d_in[7];
  const float* ln1_s  = (const float*)d_in[8];
  const float* ln1_b  = (const float*)d_in[9];
  const float* ln2_s  = (const float*)d_in[10];
  const float* ln2_b  = (const float*)d_in[11];
  const float* mlp_w1 = (const float*)d_in[12];
  const float* mlp_b1 = (const float*)d_in[13];
  const float* mlp_w2 = (const float*)d_in[14];
  const float* mlp_b2 = (const float*)d_in[15];
  const float* w_out  = (const float*)d_in[16];
  const float* b_out  = (const float*)d_in[17];

  // workspace layout (44 MB total)
  char* ws = (char*)d_ws;
  float*    x     = (float*)ws;     ws += (size_t)ROWS * EDIM * 4;       // 8 MB
  _Float16* h16   = (_Float16*)ws;  ws += (size_t)ROWS * EDIM * 2;       // 4 MB
  _Float16* qkv16 = (_Float16*)ws;  ws += (size_t)ROWS * 3 * EDIM * 2;   // 12 MB
  _Float16* o16   = (_Float16*)ws;  ws += (size_t)ROWS * EDIM * 2;       // 4 MB
  _Float16* g16   = (_Float16*)ws;  ws += (size_t)ROWS * 4 * EDIM * 2;   // 16 MB

  embed_kernel<<<ROWS, 256, 0, stream>>>(xs, ys, w_in, b_in, x);

  for (int l = 0; l < N_LAYER; ++l) {
    ln_kernel<<<ROWS, 256, 0, stream>>>(x, ln1_s + l * EDIM, ln1_b + l * EDIM,
                                        h16);
    gemm_kernel<<<dim3(3 * EDIM / 128, ROWS / 64), 256, 0, stream>>>(
        h16, qkv_w + (size_t)l * EDIM * 3 * EDIM, qkv_b + (size_t)l * 3 * EDIM,
        nullptr, nullptr, qkv16, ROWS, 3 * EDIM, EDIM, /*f16 out*/ 8);
    attn_kernel<<<BATCH * NHEAD * (LSEQ / 16), 32, 0, stream>>>(qkv16, o16);
    gemm_kernel<<<dim3(EDIM / 128, ROWS / 64), 256, 0, stream>>>(
        o16, proj_w + (size_t)l * EDIM * EDIM, proj_b + (size_t)l * EDIM,
        x, x, nullptr, ROWS, EDIM, EDIM, /*res+f32*/ 1 | 4);
    ln_kernel<<<ROWS, 256, 0, stream>>>(x, ln2_s + l * EDIM, ln2_b + l * EDIM,
                                        h16);
    gemm_kernel<<<dim3(4 * EDIM / 128, ROWS / 64), 256, 0, stream>>>(
        h16, mlp_w1 + (size_t)l * EDIM * 4 * EDIM,
        mlp_b1 + (size_t)l * 4 * EDIM, nullptr, nullptr, g16,
        ROWS, 4 * EDIM, EDIM, /*gelu+f16*/ 2 | 8);
    gemm_kernel<<<dim3(EDIM / 128, ROWS / 64), 256, 0, stream>>>(
        g16, mlp_w2 + (size_t)l * 4 * EDIM * EDIM, mlp_b2 + (size_t)l * EDIM,
        x, x, nullptr, ROWS, EDIM, 4 * EDIM, /*res+f32*/ 1 | 4);
  }

  head_kernel<<<BATCH * KSEQ, 256, 0, stream>>>(x, w_out, b_out,
                                                (float*)d_out);
}
